// StrategicReasoner_89558658056738
// MI455X (gfx1250) — compile-verified, matrix operand order checked by reading, not run on hardware
//
#include <hip/hip_runtime.h>

// ---------------------------------------------------------------------------
// MI455X (gfx1250) implementation.
// All heavy GEMMs run on v_wmma_f32_16x16x32_bf16 (bf16 in, f32 accum).
// Weights are pre-packed once per launch to transposed bf16 (N-major, K
// contiguous) so each lane's B fragment is two contiguous 16-byte loads.
// Activations stay bf16 between kernels (L2-resident at 64 MB); the running
// residual h stays fp32 for accuracy.
// GELU is a branchless tanh-form approximation on v_exp_f32 (the libm erff
// expansion creates divergent exec-mask blocks in the hot epilogues).
// ---------------------------------------------------------------------------

typedef __bf16 bf16_t;
typedef __attribute__((ext_vector_type(16))) __bf16 v16bf;
typedef __attribute__((ext_vector_type(8)))  __bf16 v8bf;
typedef __attribute__((ext_vector_type(4)))  __bf16 v4bf;
typedef __attribute__((ext_vector_type(8)))  float  v8f;
typedef __attribute__((ext_vector_type(4)))  float  v4f;

#define LN_EPS 1e-5f
#define RSQRT_DH 0.1767766953f   // 1/sqrt(32)

// Branchless GELU (tanh form): tanh(y) = 1 - 2/(exp(2y)+1)  -> v_exp_f32
__device__ __forceinline__ float gelu_f(float x) {
    float y = 0.7978845608f * (x + 0.044715f * x * x * x);
    float e = __expf(2.f * y);
    float t = 1.f - 2.f / (e + 1.f);
    return 0.5f * x * (1.f + t);
}

// ---------------------------------------------------------------------------
// Weight pack: W (K x N, fp32, row-major) -> Wt (N x K, bf16, row-major)
// ---------------------------------------------------------------------------
__global__ void pack_wt_kernel(const float* __restrict__ W,
                               bf16_t* __restrict__ Wt, int K, int N) {
    long i = (long)blockIdx.x * 256 + threadIdx.x;
    if (i >= (long)K * N) return;
    int n = (int)(i / K);
    int k = (int)(i % K);
    Wt[i] = (bf16_t)W[(long)k * N + n];
}

// ---------------------------------------------------------------------------
// LayerNorm over last dim (128): one wave per row; lane owns 4 contiguous
// columns (one b128 load, one b64 bf16 store).  Population variance.
// ---------------------------------------------------------------------------
__global__ __launch_bounds__(256)
void ln_kernel(const float* __restrict__ h, const float* __restrict__ g,
               const float* __restrict__ b, bf16_t* __restrict__ out,
               long rows) {
    long wave = ((long)blockIdx.x * blockDim.x + threadIdx.x) >> 5;
    int lane = threadIdx.x & 31;
    if (wave >= rows) return;
    v4f v0 = *(const v4f*)(h + wave * 128 + lane * 4);
    float s = v0[0] + v0[1] + v0[2] + v0[3];
#pragma unroll
    for (int m = 16; m >= 1; m >>= 1) s += __shfl_xor(s, m, 32);
    float mean = s * (1.f / 128.f);
    float vs = 0.f;
#pragma unroll
    for (int i = 0; i < 4; ++i) { float d = v0[i] - mean; vs += d * d; }
#pragma unroll
    for (int m = 16; m >= 1; m >>= 1) vs += __shfl_xor(vs, m, 32);
    float rstd = rsqrtf(vs * (1.f / 128.f) + LN_EPS);
    v4f gg = *(const v4f*)(g + lane * 4);
    v4f bb = *(const v4f*)(b + lane * 4);
    v4bf o;
#pragma unroll
    for (int i = 0; i < 4; ++i)
        o[i] = (bf16_t)((v0[i] - mean) * rstd * gg[i] + bb[i]);
    *(v4bf*)(out + wave * 128 + lane * 4) = o;
}

// ---------------------------------------------------------------------------
// Generic WMMA GEMM:  C[M, 128*grid.y .. +128) = epi(A @ Wt^T + bias [+ res])
//  - block = 128 threads = 4 waves; wave owns 16 rows x 128 cols (8 acc tiles
//    = 8 independent WMMA dependency chains per k-step)
//  - A may be split at K=splitK across two buffers (concat GEMM)
//  - TA = float (converted to bf16 on load) or bf16
//  - EPI: 0 none, 1 relu, 2 gelu
// ---------------------------------------------------------------------------
template <typename TA, typename TO, int EPI>
__global__ __launch_bounds__(128)
void gemm_wmma(const TA* __restrict__ A0, int lda0,
               const TA* __restrict__ A1, int lda1, int splitK,
               const bf16_t* __restrict__ Wt, const float* __restrict__ bias,
               const float* __restrict__ res,
               TO* __restrict__ C, int ldc, int K) {
    const int lane   = threadIdx.x & 31;
    const int wave   = threadIdx.x >> 5;
    const long rowBase = ((long)blockIdx.x * 4 + wave) * 16;
    const int colBase = blockIdx.y * 128;
    const int lr     = lane & 15;             // A row / B,C col within tile
    const int khalf8 = (lane >> 4) * 8;       // K half select per ISA layout
    const int rhalf  = (lane >> 4) * 8;       // C row half select

    v8f acc[8];
#pragma unroll
    for (int t = 0; t < 8; ++t)
#pragma unroll
        for (int j = 0; j < 8; ++j) acc[t][j] = 0.f;

    const long arow = rowBase + lr;
    for (int k0 = 0; k0 < K; k0 += 32) {
        const TA* Ap;
        if (A1 != nullptr && k0 >= splitK)
            Ap = A1 + arow * (long)lda1 + (k0 - splitK);
        else
            Ap = A0 + arow * (long)lda0 + k0;

        v16bf a;
        if constexpr (sizeof(TA) == 4) {
            const v4f* p0 = (const v4f*)((const float*)Ap + khalf8);
            v4f x0 = p0[0], x1 = p0[1];
            const v4f* p1 = (const v4f*)((const float*)Ap + 16 + khalf8);
            v4f x2 = p1[0], x3 = p1[1];
#pragma unroll
            for (int i = 0; i < 4; ++i) {
                a[i]      = (bf16_t)x0[i];
                a[4 + i]  = (bf16_t)x1[i];
                a[8 + i]  = (bf16_t)x2[i];
                a[12 + i] = (bf16_t)x3[i];
            }
        } else {
            v8bf x0 = *(const v8bf*)((const bf16_t*)Ap + khalf8);
            v8bf x1 = *(const v8bf*)((const bf16_t*)Ap + 16 + khalf8);
#pragma unroll
            for (int i = 0; i < 8; ++i) { a[i] = x0[i]; a[8 + i] = x1[i]; }
        }

#pragma unroll
        for (int nt = 0; nt < 8; ++nt) {
            const bf16_t* Bp = Wt + (long)(colBase + nt * 16 + lr) * K + k0 + khalf8;
            v8bf b0 = *(const v8bf*)Bp;
            v8bf b1 = *(const v8bf*)(Bp + 16);
            v16bf bfr;
#pragma unroll
            for (int i = 0; i < 8; ++i) { bfr[i] = b0[i]; bfr[8 + i] = b1[i]; }
            acc[nt] = __builtin_amdgcn_wmma_f32_16x16x32_bf16(
                false, a, false, bfr, (short)0, acc[nt], false, false);
        }
    }

#pragma unroll
    for (int nt = 0; nt < 8; ++nt) {
        int c = colBase + nt * 16 + lr;
        float bv = bias ? bias[c] : 0.f;
#pragma unroll
        for (int j = 0; j < 8; ++j) {
            long rr = rowBase + rhalf + j;
            float v = acc[nt][j] + bv;
            if (res) v += res[rr * (long)ldc + c];
            if constexpr (EPI == 1) v = v > 0.f ? v : 0.f;
            if constexpr (EPI == 2) v = gelu_f(v);
            C[rr * (long)ldc + c] = (TO)v;
        }
    }
}

// ---------------------------------------------------------------------------
// Tiny causal MHA core: one thread per (b, head, node).  N=8, DH=32.
// q,k,v,att all bf16 (B*N, 128), head hh uses columns [hh*32, hh*32+32).
// ---------------------------------------------------------------------------
__global__ __launch_bounds__(256)
void attn_kernel(const bf16_t* __restrict__ q, const bf16_t* __restrict__ k,
                 const bf16_t* __restrict__ v, bf16_t* __restrict__ att) {
    long idx = (long)blockIdx.x * blockDim.x + threadIdx.x;
    int n  = (int)(idx & 7);  idx >>= 3;
    int hh = (int)(idx & 3);  idx >>= 2;
    long b = idx;
    if (b >= 32768) return;
    const long base = b * 8 * 128 + hh * 32;

    float qv[32];
    {
        const v8bf* qp = (const v8bf*)(q + base + (long)n * 128);
#pragma unroll
        for (int c = 0; c < 4; ++c) {
            v8bf t = qp[c];
#pragma unroll
            for (int i = 0; i < 8; ++i) qv[c * 8 + i] = (float)t[i];
        }
    }

    float s[8];
    float mx = -1e30f;
#pragma unroll
    for (int m = 0; m < 8; ++m) {
        if (m <= n) {
            const v8bf* kp = (const v8bf*)(k + base + (long)m * 128);
            float acc = 0.f;
#pragma unroll
            for (int c = 0; c < 4; ++c) {
                v8bf t = kp[c];
#pragma unroll
                for (int i = 0; i < 8; ++i) acc += qv[c * 8 + i] * (float)t[i];
            }
            acc *= RSQRT_DH;
            s[m] = acc;
            mx = fmaxf(mx, acc);
        }
    }
    float denom = 0.f;
#pragma unroll
    for (int m = 0; m < 8; ++m) {
        if (m <= n) { s[m] = __expf(s[m] - mx); denom += s[m]; }
    }
    float inv = 1.f / denom;
    float out[32];
#pragma unroll
    for (int d = 0; d < 32; ++d) out[d] = 0.f;
#pragma unroll
    for (int m = 0; m < 8; ++m) {
        if (m <= n) {
            float a = s[m] * inv;
            const v8bf* vp = (const v8bf*)(v + base + (long)m * 128);
#pragma unroll
            for (int c = 0; c < 4; ++c) {
                v8bf t = vp[c];
#pragma unroll
                for (int i = 0; i < 8; ++i) out[c * 8 + i] += a * (float)t[i];
            }
        }
    }
#pragma unroll
    for (int c = 0; c < 4; ++c) {
        v8bf o;
#pragma unroll
        for (int i = 0; i < 8; ++i) o[i] = (bf16_t)out[c * 8 + i];
        *(v8bf*)(att + base + (long)n * 128 + c * 8) = o;
    }
}

// ---------------------------------------------------------------------------
// Fused FFN: h += gelu(hn @ Wf1 + b1) @ Wf2 + b2.
// Stage 1 processes 4 n-tiles at a time (4 independent WMMA chains) and
// writes the 64x512 gelu intermediate to LDS (bf16, padded stride); stage 2
// streams it back as WMMA A-fragments.  The fp32 intermediate never touches
// HBM.
// ---------------------------------------------------------------------------
__global__ __launch_bounds__(128)
void ffn_kernel(const bf16_t* __restrict__ hn,
                const bf16_t* __restrict__ W1t,  // (512 x 128) bf16
                const float* __restrict__ b1,    // 512
                const bf16_t* __restrict__ W2t,  // (128 x 512) bf16
                const float* __restrict__ b2,    // 128
                float* __restrict__ h) {         // (R x 128) residual in-place
    extern __shared__ bf16_t lds[];              // 64 rows x 520 (pad 8)
    const int LSTR = 520;
    const int lane   = threadIdx.x & 31;
    const int wave   = threadIdx.x >> 5;
    const int lr     = lane & 15;
    const int khalf8 = (lane >> 4) * 8;
    const int rhalf  = (lane >> 4) * 8;
    const long rowBase = (long)blockIdx.x * 64 + wave * 16;

    // preload A fragments for K=128 (4 k-steps)
    v16bf afrag[4];
#pragma unroll
    for (int ks = 0; ks < 4; ++ks) {
        const bf16_t* Ap = hn + (rowBase + lr) * 128 + ks * 32 + khalf8;
        v8bf x0 = *(const v8bf*)Ap;
        v8bf x1 = *(const v8bf*)(Ap + 16);
#pragma unroll
        for (int i = 0; i < 8; ++i) { afrag[ks][i] = x0[i]; afrag[ks][8 + i] = x1[i]; }
    }

    // stage 1: t = gelu(hn @ Wf1 + b1) -> LDS, 4 n-tiles per pass
    for (int ntg = 0; ntg < 8; ++ntg) {
        v8f acc[4];
#pragma unroll
        for (int u = 0; u < 4; ++u)
#pragma unroll
            for (int j = 0; j < 8; ++j) acc[u][j] = 0.f;
#pragma unroll
        for (int ks = 0; ks < 4; ++ks) {
#pragma unroll
            for (int u = 0; u < 4; ++u) {
                const bf16_t* Bp =
                    W1t + (long)((ntg * 4 + u) * 16 + lr) * 128 + ks * 32 + khalf8;
                v8bf b0 = *(const v8bf*)Bp;
                v8bf b1v = *(const v8bf*)(Bp + 16);
                v16bf bfr;
#pragma unroll
                for (int i = 0; i < 8; ++i) { bfr[i] = b0[i]; bfr[8 + i] = b1v[i]; }
                acc[u] = __builtin_amdgcn_wmma_f32_16x16x32_bf16(
                    false, afrag[ks], false, bfr, (short)0, acc[u], false, false);
            }
        }
#pragma unroll
        for (int u = 0; u < 4; ++u) {
            int c = (ntg * 4 + u) * 16 + lr;
            float bb = b1[c];
#pragma unroll
            for (int j = 0; j < 8; ++j) {
                float x = gelu_f(acc[u][j] + bb);
                lds[(wave * 16 + rhalf + j) * LSTR + c] = (bf16_t)x;
            }
        }
    }
    __syncthreads();

    // stage 2: h += t @ Wf2 + b2  (K=512 from LDS, 8 independent chains)
    v8f acc2[8];
#pragma unroll
    for (int t = 0; t < 8; ++t)
#pragma unroll
        for (int j = 0; j < 8; ++j) acc2[t][j] = 0.f;

    for (int ks = 0; ks < 16; ++ks) {
        const bf16_t* Ap = lds + (wave * 16 + lr) * LSTR + ks * 32 + khalf8;
        v8bf x0 = *(const v8bf*)Ap;
        v8bf x1 = *(const v8bf*)(Ap + 16);
        v16bf a;
#pragma unroll
        for (int i = 0; i < 8; ++i) { a[i] = x0[i]; a[8 + i] = x1[i]; }
#pragma unroll
        for (int nt = 0; nt < 8; ++nt) {
            const bf16_t* Bp = W2t + (long)(nt * 16 + lr) * 512 + ks * 32 + khalf8;
            v8bf b0 = *(const v8bf*)Bp;
            v8bf b1v = *(const v8bf*)(Bp + 16);
            v16bf bfr;
#pragma unroll
            for (int i = 0; i < 8; ++i) { bfr[i] = b0[i]; bfr[8 + i] = b1v[i]; }
            acc2[nt] = __builtin_amdgcn_wmma_f32_16x16x32_bf16(
                false, a, false, bfr, (short)0, acc2[nt], false, false);
        }
    }
#pragma unroll
    for (int nt = 0; nt < 8; ++nt) {
        int c = nt * 16 + lr;
        float bb = b2[c];
#pragma unroll
        for (int j = 0; j < 8; ++j) {
            long rr = rowBase + rhalf + j;
            h[rr * 128 + c] += acc2[nt][j] + bb;
        }
    }
}

// ---------------------------------------------------------------------------
// Launcher
// ---------------------------------------------------------------------------
extern "C" void kernel_launch(void* const* d_in, const int* in_sizes, int n_in,
                              void* d_out, int out_size, void* d_ws, size_t ws_size,
                              hipStream_t stream) {
    const long Bsz = 32768, Nn = 8, H = 128;
    const long R = Bsz * Nn;               // 262144 token rows
    const int NOSPLIT = 1 << 30;

    const float* x     = (const float*)d_in[0];
    const float* Wp    = (const float*)d_in[1];
    const float* bp    = (const float*)d_in[2];
    const float* ln1_g = (const float*)d_in[3];
    const float* ln1_b = (const float*)d_in[4];
    const float* Wq = (const float*)d_in[5];  const float* bq = (const float*)d_in[6];
    const float* Wk = (const float*)d_in[7];  const float* bk = (const float*)d_in[8];
    const float* Wv = (const float*)d_in[9];  const float* bv = (const float*)d_in[10];
    const float* Wo = (const float*)d_in[11]; const float* bo = (const float*)d_in[12];
    const float* Wu = (const float*)d_in[13]; const float* bu = (const float*)d_in[14];
    const float* ln2_g = (const float*)d_in[15]; const float* ln2_b = (const float*)d_in[16];
    const float* Wf1 = (const float*)d_in[17]; const float* bf1 = (const float*)d_in[18];
    const float* Wf2 = (const float*)d_in[19]; const float* bf2 = (const float*)d_in[20];
    const float* Wa  = (const float*)d_in[21]; const float* ba  = (const float*)d_in[22];
    const float* Wout = (const float*)d_in[23]; const float* bout = (const float*)d_in[24];

    // workspace partition
    char* ws = (char*)d_ws;
    size_t off = 0;
    auto alloc = [&](size_t bytes) -> char* {
        char* p = ws + off;
        off = (off + bytes + 255) & ~(size_t)255;
        return p;
    };
    float*  h    = (float*) alloc(R * H * 4);   // running residual (fp32)
    bf16_t* hn   = (bf16_t*)alloc(R * H * 2);   // LN output / GEMM A
    bf16_t* qb   = (bf16_t*)alloc(R * H * 2);
    bf16_t* kb   = (bf16_t*)alloc(R * H * 2);
    bf16_t* vb   = (bf16_t*)alloc(R * H * 2);
    bf16_t* attb = (bf16_t*)alloc(R * H * 2);
    float*  aggb = (float*) alloc(Bsz * H * 4);
    bf16_t* wWp   = (bf16_t*)alloc((size_t)256 * 1024 * 2);
    bf16_t* wWq   = (bf16_t*)alloc((size_t)3 * 128 * 128 * 2);
    bf16_t* wWk   = (bf16_t*)alloc((size_t)3 * 128 * 128 * 2);
    bf16_t* wWv   = (bf16_t*)alloc((size_t)3 * 128 * 128 * 2);
    bf16_t* wWo   = (bf16_t*)alloc((size_t)3 * 128 * 128 * 2);
    bf16_t* wWu   = (bf16_t*)alloc((size_t)3 * 256 * 128 * 2);
    bf16_t* wWf1  = (bf16_t*)alloc((size_t)3 * 128 * 512 * 2);
    bf16_t* wWf2  = (bf16_t*)alloc((size_t)3 * 512 * 128 * 2);
    bf16_t* wWa   = (bf16_t*)alloc((size_t)1024 * 128 * 2);
    bf16_t* wWout = (bf16_t*)alloc((size_t)128 * 128 * 2);
    bf16_t* atto = qb;   // reuse q buffer after attention

    auto packW = [&](const float* W, bf16_t* Wt, int K, int N) {
        long total = (long)K * N;
        pack_wt_kernel<<<dim3((unsigned)((total + 255) / 256)), dim3(256), 0, stream>>>(W, Wt, K, N);
    };

    // ---- pack all weights to transposed bf16 ----
    packW(Wp, wWp, 256, 1024);
    for (int l = 0; l < 3; ++l) {
        packW(Wq + l * 128 * 128, wWq + l * 128 * 128, 128, 128);
        packW(Wk + l * 128 * 128, wWk + l * 128 * 128, 128, 128);
        packW(Wv + l * 128 * 128, wWv + l * 128 * 128, 128, 128);
        packW(Wo + l * 128 * 128, wWo + l * 128 * 128, 128, 128);
        packW(Wu + l * 256 * 128, wWu + l * 256 * 128, 256, 128);
        packW(Wf1 + l * 128 * 512, wWf1 + l * 128 * 512, 128, 512);
        packW(Wf2 + l * 512 * 128, wWf2 + l * 512 * 128, 512, 128);
    }
    packW(Wa, wWa, 1024, 128);
    packW(Wout, wWout, 128, 128);

    const dim3 blk128(128), blk256(256);
    const unsigned gTok = (unsigned)(R / 64);       // 4096 row tiles of 64
    const unsigned gB   = (unsigned)(Bsz / 64);     // 512

    // ---- input projection: h = x @ Wp + bp  (M=32768, K=256, N=1024) ----
    gemm_wmma<float, float, 0><<<dim3(gB, 8), blk128, 0, stream>>>(
        x, 256, (const float*)nullptr, 0, NOSPLIT, wWp, bp, nullptr, h, 1024, 256);

    for (int l = 0; l < 3; ++l) {
        const bf16_t* wq = wWq + l * 128 * 128;
        const bf16_t* wk = wWk + l * 128 * 128;
        const bf16_t* wv = wWv + l * 128 * 128;
        const bf16_t* wo = wWo + l * 128 * 128;
        const bf16_t* wu = wWu + l * 256 * 128;
        const bf16_t* w1 = wWf1 + l * 128 * 512;
        const bf16_t* w2 = wWf2 + l * 512 * 128;

        // LN1 -> hn (bf16)
        ln_kernel<<<dim3((unsigned)(R / 8)), blk256, 0, stream>>>(
            h, ln1_g + l * 128, ln1_b + l * 128, hn, R);

        // Q, K, V projections
        gemm_wmma<bf16_t, bf16_t, 0><<<dim3(gTok, 1), blk128, 0, stream>>>(
            hn, 128, (const bf16_t*)nullptr, 0, NOSPLIT, wq, bq + l * 128, nullptr, qb, 128, 128);
        gemm_wmma<bf16_t, bf16_t, 0><<<dim3(gTok, 1), blk128, 0, stream>>>(
            hn, 128, (const bf16_t*)nullptr, 0, NOSPLIT, wk, bk + l * 128, nullptr, kb, 128, 128);
        gemm_wmma<bf16_t, bf16_t, 0><<<dim3(gTok, 1), blk128, 0, stream>>>(
            hn, 128, (const bf16_t*)nullptr, 0, NOSPLIT, wv, bv + l * 128, nullptr, vb, 128, 128);

        // causal MHA core (tiny; VALU)
        attn_kernel<<<dim3((unsigned)(Bsz * 4 * 8 / 256)), blk256, 0, stream>>>(qb, kb, vb, attb);

        // output projection: atto = att @ Wo + bo   (reuses q buffer)
        gemm_wmma<bf16_t, bf16_t, 0><<<dim3(gTok, 1), blk128, 0, stream>>>(
            attb, 128, (const bf16_t*)nullptr, 0, NOSPLIT, wo, bo + l * 128, nullptr, atto, 128, 128);

        // update net: h = relu(h + cat([hn, atto]) @ Wu + bu)  (split-K GEMM)
        gemm_wmma<bf16_t, float, 1><<<dim3(gTok, 1), blk128, 0, stream>>>(
            hn, 128, atto, 128, 128, wu, bu + l * 128, h, h, 128, 256);

        // LN2 -> hn
        ln_kernel<<<dim3((unsigned)(R / 8)), blk256, 0, stream>>>(
            h, ln2_g + l * 128, ln2_b + l * 128, hn, R);

        // fused FFN: h += gelu(hn @ Wf1 + b1) @ Wf2 + b2
        ffn_kernel<<<dim3(gTok), blk128, 64 * 520 * 2, stream>>>(
            hn, w1, bf1 + l * 512, w2, bf2 + l * 128, h);
    }

    // ---- aggregate: agg = gelu(h.reshape(B,1024) @ Wa + ba) ----
    gemm_wmma<float, float, 2><<<dim3(gB, 1), blk128, 0, stream>>>(
        h, 1024, (const float*)nullptr, 0, NOSPLIT, wWa, ba, nullptr, aggb, 128, 1024);

    // ---- head: out = agg @ Wout + bout ----
    gemm_wmma<float, float, 0><<<dim3(gB, 1), blk128, 0, stream>>>(
        aggb, 128, (const float*)nullptr, 0, NOSPLIT, wWout, bout, nullptr,
        (float*)d_out, 128, 128);
}
